// TransformerEdgeClassifier_28226525069858
// MI455X (gfx1250) — matrix operand order repaired and checked by reading, not compiled, not run on hardware
//
#include <hip/hip_runtime.h>

// ---------------------------------------------------------------------------
// TransformerEdgeClassifier for MI455X (gfx1250, wave32, WMMA + TDM)
// ---------------------------------------------------------------------------
#define NN 20000
#define NE 320000
#define CDIV(a, b) (((a) + (b) - 1) / (b))

typedef __attribute__((ext_vector_type(16))) __bf16 v16bf;
typedef __attribute__((ext_vector_type(8)))  __bf16 v8bf;
typedef __attribute__((ext_vector_type(8)))  float  v8f;
typedef __attribute__((ext_vector_type(4)))  unsigned int u32x4;
typedef __attribute__((ext_vector_type(8)))  int i32x8;
typedef __attribute__((ext_vector_type(4)))  int i32x4;

// ---------------------------------------------------------------------------
// Tensor Data Mover: DMA a [rows x elems] fp32 2D tile (row stride = elems)
// from global memory into LDS.  D# per CDNA5 ISA §8.3/8.4:
//   group0: count=1 | lds_addr | global_addr[31:0] | global_addr[56:32],type=2
//   group1: data_size=4B, tensor_dim0/1, tile_dim0/1, dim0_stride
// Issue from ONE wave; consumer waits s_wait_tensorcnt==0 then block barrier.
// This toolchain's builtin takes 6 args (g0, g1, g2, g3, g4, cpol).
// ---------------------------------------------------------------------------
__device__ __forceinline__ void tdm_load_rows_to_lds(void* lds_ptr, const float* gsrc,
                                                     unsigned elems, unsigned rows) {
    unsigned lds_off = (unsigned)(unsigned long long)lds_ptr;  // flat[31:0] == LDS offset
    unsigned long long ga = (unsigned long long)gsrc;
    u32x4 g0;
    g0[0] = 1u;                                                 // count=1, user descriptor
    g0[1] = lds_off;
    g0[2] = (unsigned)ga;
    g0[3] = ((unsigned)(ga >> 32) & 0x1ffffffu) | (2u << 30);   // addr[56:32] | type=2
    i32x8 g1;
    g1[0] = (int)(2u << 16);                                    // data_size=4B, no multicast
    g1[1] = (int)((elems & 0xffffu) << 16);                     // tensor_dim0 lo16 @bits63:48
    g1[2] = (int)(((elems >> 16) & 0xffffu) | ((rows & 0xffffu) << 16)); // dim0 hi | dim1 lo
    g1[3] = (int)(((rows >> 16) & 0xffffu) | ((elems & 0xffffu) << 16)); // dim1 hi | tile_dim0
    g1[4] = (int)(rows & 0xffffu);                              // tile_dim1 (tile_dim2=0 -> 2D)
    g1[5] = (int)elems;                                         // tensor_dim0_stride lo32
    g1[6] = 0;
    g1[7] = 0;
    i32x4 gz4 = {0, 0, 0, 0};
    i32x8 gz8 = {0, 0, 0, 0, 0, 0, 0, 0};
    __builtin_amdgcn_tensor_load_to_lds(g0, g1, gz4, gz4, gz8, 0);
}

// ---------------------------------------------------------------------------
// Prep kernels
// ---------------------------------------------------------------------------
__global__ void k_wt_convert(const float* __restrict__ src, __bf16* __restrict__ dst,
                             int K, int Nn) {
    int idx = blockIdx.x * blockDim.x + threadIdx.x;
    if (idx >= K * Nn) return;
    int n = idx / K, k = idx % K;
    dst[idx] = (__bf16)src[(size_t)k * Nn + n];
}

__global__ void k_copy_f32(const float* __restrict__ src, float* __restrict__ dst, int n) {
    int i = blockIdx.x * blockDim.x + threadIdx.x;
    if (i < n) dst[i] = src[i];
}

// enc_table[t][j] = sum_d type_emb[t][d] * enc_w[d][j] + enc_b[j]   (16x64, tiny)
__global__ void k_enc_table(const float* __restrict__ te, const float* __restrict__ ew,
                            const float* __restrict__ eb, float* __restrict__ tab) {
    int idx = blockIdx.x * blockDim.x + threadIdx.x;
    if (idx >= 16 * 64) return;
    int t = idx >> 6, j = idx & 63;
    float a = eb[j];
    #pragma unroll 8
    for (int d = 0; d < 64; ++d) a += te[t * 64 + d] * ew[d * 64 + j];
    tab[idx] = a;
}

// h0 (bf16) = enc_table[x[n]]
__global__ void k_encode(const int* __restrict__ x, const float* __restrict__ tab,
                         __bf16* __restrict__ hbf) {
    int idx = blockIdx.x * blockDim.x + threadIdx.x;
    if (idx >= NN * 64) return;
    int n = idx >> 6, j = idx & 63;
    hbf[idx] = (__bf16)tab[(x[n] & 15) * 64 + j];
}

// m = -inf, z = 0, acc = 0
__global__ void k_init(float* __restrict__ m, float* __restrict__ z,
                       float* __restrict__ acc, int nh, int nhc) {
    int i = blockIdx.x * blockDim.x + threadIdx.x;
    if (i < nh) { m[i] = -1e30f; z[i] = 0.f; }
    if (i < nhc) acc[i] = 0.f;
}

// ---------------------------------------------------------------------------
// GEMM: [M,K](bf16) x WT[Ntot,K](bf16, pre-transposed) + bias
// One wave = one 16x16 tile via v_wmma_f32_16x16x32_bf16 (layouts per ISA 7.12.2).
// Epilogue splits: cols [0,nqkv) -> bf16 qkv buffer (stride nqkv);
//                  cols [nqkv,Ntot) -> fp32 skip buffer (stride Ntot-nqkv).
// ---------------------------------------------------------------------------
template <int K>
__global__ __launch_bounds__(128) void k_gemm(const __bf16* __restrict__ A,
                                              const __bf16* __restrict__ WT,
                                              const float* __restrict__ bias,
                                              __bf16* __restrict__ Cq,
                                              float* __restrict__ Cs,
                                              int M, int Ntot, int nqkv) {
    int lane = threadIdx.x & 31;
    int wid  = blockIdx.x * (blockDim.x >> 5) + (threadIdx.x >> 5);
    int ntiles = Ntot >> 4;
    if (wid >= (M >> 4) * ntiles) return;
    int nt = wid % ntiles, mt = wid / ntiles;
    int half = lane >> 4, lr = lane & 15;

    const __bf16* arow = A  + (size_t)(mt * 16 + lr) * K + half * 8;
    const __bf16* brow = WT + (size_t)(nt * 16 + lr) * K + half * 8;

    v8f acc = {0.f, 0.f, 0.f, 0.f, 0.f, 0.f, 0.f, 0.f};
    #pragma unroll
    for (int k0 = 0; k0 < K; k0 += 32) {
        if (k0 + 32 < K) __builtin_prefetch(arow + k0 + 32, 0, 1);
        v8bf alo = *(const v8bf*)(arow + k0);
        v8bf ahi = *(const v8bf*)(arow + k0 + 16);
        v8bf blo = *(const v8bf*)(brow + k0);
        v8bf bhi = *(const v8bf*)(brow + k0 + 16);
        v16bf a, b;
        #pragma unroll
        for (int i = 0; i < 8; ++i) {
            a[i] = alo[i]; a[i + 8] = ahi[i];
            b[i] = blo[i]; b[i + 8] = bhi[i];
        }
        acc = __builtin_amdgcn_wmma_f32_16x16x32_bf16(false, a, false, b,
                                                      (short)0, acc, false, false);
    }
    int col = nt * 16 + lr;
    float bv = bias[col];
    if (col < nqkv) {
        #pragma unroll
        for (int r = 0; r < 8; ++r)
            Cq[(size_t)(mt * 16 + r + half * 8) * nqkv + col] = (__bf16)(acc[r] + bv);
    } else {
        int scol = col - nqkv, sw = Ntot - nqkv;
        #pragma unroll
        for (int r = 0; r < 8; ++r)
            Cs[(size_t)(mt * 16 + r + half * 8) * sw + scol] = acc[r] + bv;
    }
}

// ---------------------------------------------------------------------------
// Edge pass A: scores + segment max.  One wave per edge.  C == 64 assumed.
// qkv (bf16) row layout: [q(0..HC) | k(HC..2HC) | v(2HC..3HC)], stride 3*HC.
// `we` [3,HC] staged into LDS by the Tensor Data Mover.
// ---------------------------------------------------------------------------
template <int H, int HC>
__global__ __launch_bounds__(256) void k_edge_scores(const int* __restrict__ ei,
                                                     const float* __restrict__ ea,
                                                     const float* __restrict__ we,
                                                     const __bf16* __restrict__ qkv,
                                                     float* __restrict__ scores,
                                                     float* __restrict__ mbuf) {
    __shared__ float swe[3 * HC];
    if (threadIdx.x < 32) {
        tdm_load_rows_to_lds(swe, we, HC, 3);
        __builtin_amdgcn_s_wait_tensorcnt(0);
    }
    __syncthreads();

    int lane = threadIdx.x & 31;
    int e = blockIdx.x * (blockDim.x >> 5) + (threadIdx.x >> 5);
    if (e >= NE) return;
    int src = ei[e], dst = ei[NE + e];
    float a0 = ea[e * 3 + 0], a1 = ea[e * 3 + 1], a2 = ea[e * 3 + 2];
    const __bf16* qrow = qkv + (size_t)dst * (3 * HC);
    const __bf16* krow = qkv + (size_t)src * (3 * HC) + HC;

    float hacc[H];
    #pragma unroll
    for (int h = 0; h < H; ++h) hacc[h] = 0.f;
    #pragma unroll
    for (int t = 0; t < HC / 32; ++t) {
        int j = t * 32 + lane;
        float ef = a0 * swe[j] + a1 * swe[HC + j] + a2 * swe[2 * HC + j];
        hacc[t >> 1] += (float)qrow[j] * ((float)krow[j] + ef);   // head = j/64 = t/2
    }
    #pragma unroll
    for (int h = 0; h < H; ++h) {
        float s = hacc[h];
        #pragma unroll
        for (int off = 16; off > 0; off >>= 1) s += __shfl_xor(s, off, 32);
        s *= 0.125f;                                               // 1/sqrt(64)
        if (lane == 0) {
            scores[(size_t)e * H + h] = s;
            __hip_atomic_fetch_max(&mbuf[(size_t)dst * H + h], s,
                                   __ATOMIC_RELAXED, __HIP_MEMORY_SCOPE_AGENT);
        }
    }
}

// Edge pass B: exp, z-accumulate, unnormalized message scatter-add.
template <int H, int HC>
__global__ __launch_bounds__(256) void k_edge_accum(const int* __restrict__ ei,
                                                    const float* __restrict__ ea,
                                                    const float* __restrict__ we,
                                                    const __bf16* __restrict__ qkv,
                                                    const float* __restrict__ scores,
                                                    const float* __restrict__ mbuf,
                                                    float* __restrict__ zbuf,
                                                    float* __restrict__ accbuf) {
    __shared__ float swe[3 * HC];
    if (threadIdx.x < 32) {
        tdm_load_rows_to_lds(swe, we, HC, 3);
        __builtin_amdgcn_s_wait_tensorcnt(0);
    }
    __syncthreads();

    int lane = threadIdx.x & 31;
    int e = blockIdx.x * (blockDim.x >> 5) + (threadIdx.x >> 5);
    if (e >= NE) return;
    int src = ei[e], dst = ei[NE + e];
    float a0 = ea[e * 3 + 0], a1 = ea[e * 3 + 1], a2 = ea[e * 3 + 2];

    float exh[H];
    #pragma unroll
    for (int h = 0; h < H; ++h) {
        exh[h] = __expf(scores[(size_t)e * H + h] - mbuf[(size_t)dst * H + h]);
        if (lane == h)
            __hip_atomic_fetch_add(&zbuf[(size_t)dst * H + h], exh[h],
                                   __ATOMIC_RELAXED, __HIP_MEMORY_SCOPE_AGENT);
    }
    const __bf16* vrow = qkv + (size_t)src * (3 * HC) + 2 * HC;
    #pragma unroll
    for (int t = 0; t < HC / 32; ++t) {
        int j = t * 32 + lane;
        float ef = a0 * swe[j] + a1 * swe[HC + j] + a2 * swe[2 * HC + j];
        __hip_atomic_fetch_add(&accbuf[(size_t)dst * HC + j],
                               exh[t >> 1] * ((float)vrow[j] + ef),
                               __ATOMIC_RELAXED, __HIP_MEMORY_SCOPE_AGENT);
    }
}

// Node epilogue: normalize, +skip, LayerNorm, ReLU, (+residual).
// Writes fp32 h (for residual) AND bf16 h (next layer's WMMA operand).
template <int H, int HC, bool RES>
__global__ __launch_bounds__(256) void k_node_post(const float* __restrict__ skipb,
                                                   const float* __restrict__ accbuf,
                                                   const float* __restrict__ zbuf,
                                                   const float* __restrict__ lnw,
                                                   const float* __restrict__ lnb,
                                                   const float* __restrict__ hin,
                                                   float* __restrict__ hout,
                                                   __bf16* __restrict__ houtbf) {
    int lane = threadIdx.x & 31;
    int n = blockIdx.x * (blockDim.x >> 5) + (threadIdx.x >> 5);
    if (n >= NN) return;

    float rz[H];
    #pragma unroll
    for (int h = 0; h < H; ++h) {
        float z = zbuf[(size_t)n * H + h];
        rz[h] = (z > 0.f) ? (1.f / z) : 0.f;
    }
    float vals[HC / 32];
    float sum = 0.f;
    #pragma unroll
    for (int t = 0; t < HC / 32; ++t) {
        int j = t * 32 + lane;
        float v = accbuf[(size_t)n * HC + j] * rz[t >> 1] + skipb[(size_t)n * HC + j];
        vals[t] = v;
        sum += v;
    }
    #pragma unroll
    for (int off = 16; off > 0; off >>= 1) sum += __shfl_xor(sum, off, 32);
    float mu = sum * (1.f / (float)HC);
    float var = 0.f;
    #pragma unroll
    for (int t = 0; t < HC / 32; ++t) { float d = vals[t] - mu; var += d * d; }
    #pragma unroll
    for (int off = 16; off > 0; off >>= 1) var += __shfl_xor(var, off, 32);
    float rs = rsqrtf(var * (1.f / (float)HC) + 1e-5f);
    #pragma unroll
    for (int t = 0; t < HC / 32; ++t) {
        int j = t * 32 + lane;
        float o = (vals[t] - mu) * rs * lnw[j] + lnb[j];
        o = fmaxf(o, 0.f);
        if (RES) o += hin[(size_t)n * HC + j];
        hout[(size_t)n * HC + j]   = o;
        houtbf[(size_t)n * HC + j] = (__bf16)o;
    }
}

// ---------------------------------------------------------------------------
// Edge MLP: out[e] = relu([h[src]||h[dst]] @ w1 + b1) @ w2 + b2
// Wave per 16-edge tile; stage 1 via WMMA ([16,128]x[128,64]) on bf16 h,
// stage 2 (x[64,3]) via in-register cross-lane reduction.
// ---------------------------------------------------------------------------
__global__ __launch_bounds__(256) void k_mlp(const int* __restrict__ ei,
                                             const __bf16* __restrict__ hbf,   // [NN][64]
                                             const __bf16* __restrict__ WTm,   // [64][128]
                                             const float* __restrict__ b1,
                                             const float* __restrict__ w2,     // [64][3]
                                             const float* __restrict__ b2,
                                             float* __restrict__ out) {        // [NE][3]
    int lane = threadIdx.x & 31;
    int tile = blockIdx.x * (blockDim.x >> 5) + (threadIdx.x >> 5);
    if (tile >= NE / 16) return;
    int half = lane >> 4, lr = lane & 15;
    int e = tile * 16 + lr;
    const __bf16* hs = hbf + (size_t)ei[e] * 64;
    const __bf16* hd = hbf + (size_t)ei[NE + e] * 64;

    v8f acc[4];
    #pragma unroll
    for (int nt = 0; nt < 4; ++nt) acc[nt] = (v8f){0.f,0.f,0.f,0.f,0.f,0.f,0.f,0.f};

    #pragma unroll
    for (int k0 = 0; k0 < 128; k0 += 32) {
        int j0 = k0 + half * 8;
        int j1 = j0 + 16;
        const __bf16* p0 = (j0 < 64) ? (hs + j0) : (hd + (j0 - 64));
        const __bf16* p1 = (j1 < 64) ? (hs + j1) : (hd + (j1 - 64));
        v8bf alo = *(const v8bf*)p0;
        v8bf ahi = *(const v8bf*)p1;
        v16bf a;
        #pragma unroll
        for (int i = 0; i < 8; ++i) { a[i] = alo[i]; a[i + 8] = ahi[i]; }
        #pragma unroll
        for (int nt = 0; nt < 4; ++nt) {
            const __bf16* brow = WTm + (size_t)(nt * 16 + lr) * 128 + k0 + half * 8;
            v8bf blo = *(const v8bf*)brow;
            v8bf bhi = *(const v8bf*)(brow + 16);
            v16bf b;
            #pragma unroll
            for (int i = 0; i < 8; ++i) { b[i] = blo[i]; b[i + 8] = bhi[i]; }
            acc[nt] = __builtin_amdgcn_wmma_f32_16x16x32_bf16(false, a, false, b,
                                                              (short)0, acc[nt],
                                                              false, false);
        }
    }
    float b1v[4], w2v[4][3];
    #pragma unroll
    for (int nt = 0; nt < 4; ++nt) {
        int col = nt * 16 + lr;
        b1v[nt] = b1[col];
        w2v[nt][0] = w2[col * 3 + 0];
        w2v[nt][1] = w2[col * 3 + 1];
        w2v[nt][2] = w2[col * 3 + 2];
    }
    float bb0 = b2[0], bb1 = b2[1], bb2 = b2[2];
    #pragma unroll
    for (int r = 0; r < 8; ++r) {
        float p0 = 0.f, p1 = 0.f, p2 = 0.f;
        #pragma unroll
        for (int nt = 0; nt < 4; ++nt) {
            float h2 = fmaxf(acc[nt][r] + b1v[nt], 0.f);
            p0 += h2 * w2v[nt][0];
            p1 += h2 * w2v[nt][1];
            p2 += h2 * w2v[nt][2];
        }
        #pragma unroll
        for (int off = 8; off > 0; off >>= 1) {   // reduce within each 16-lane half
            p0 += __shfl_xor(p0, off, 32);
            p1 += __shfl_xor(p1, off, 32);
            p2 += __shfl_xor(p2, off, 32);
        }
        if (lr == 0) {
            size_t row = (size_t)(tile * 16 + r + half * 8);
            out[row * 3 + 0] = p0 + bb0;
            out[row * 3 + 1] = p1 + bb1;
            out[row * 3 + 2] = p2 + bb2;
        }
    }
}

// ---------------------------------------------------------------------------
// Host launch
// ---------------------------------------------------------------------------
extern "C" void kernel_launch(void* const* d_in, const int* in_sizes, int n_in,
                              void* d_out, int out_size, void* d_ws, size_t ws_size,
                              hipStream_t stream) {
    (void)in_sizes; (void)n_in; (void)out_size; (void)ws_size;
    const int*   x    = (const int*)d_in[0];
    const int*   ei   = (const int*)d_in[1];
    const float* ea   = (const float*)d_in[2];
    const float* te   = (const float*)d_in[3];
    const float* encw = (const float*)d_in[4];
    const float* encb = (const float*)d_in[5];
    // p1: 6..16, p2: 17..27, p3: 28..38  (wq,bq,wk,bk,wv,bv,we,ws,bs,lnw,lnb)
    const float* w1 = (const float*)d_in[39];
    const float* b1 = (const float*)d_in[40];
    const float* w2 = (const float*)d_in[41];
    const float* b2 = (const float*)d_in[42];
    float* out = (float*)d_out;

    char* wp = (char*)d_ws;
    auto alloc = [&](size_t bytes) -> char* {
        char* p = wp;
        wp += (bytes + 255) & ~(size_t)255;
        return p;
    };
    __bf16* WT1   = (__bf16*)alloc((size_t)1024 * 64 * 2);
    __bf16* WT2   = (__bf16*)alloc((size_t)1024 * 256 * 2);
    __bf16* WT3   = (__bf16*)alloc((size_t)256 * 256 * 2);
    __bf16* WTm   = (__bf16*)alloc((size_t)64 * 128 * 2);
    float*  bia1  = (float*)alloc(1024 * 4);
    float*  bia2  = (float*)alloc(1024 * 4);
    float*  bia3  = (float*)alloc(256 * 4);
    float*  tab   = (float*)alloc(16 * 64 * 4);
    float*  hA    = (float*)alloc((size_t)NN * 256 * 4);   // fp32 h (residual chain)
    float*  hB    = (float*)alloc((size_t)NN * 256 * 4);
    __bf16* hbf   = (__bf16*)alloc((size_t)NN * 256 * 2);  // bf16 h (WMMA operand)
    __bf16* qkvb  = (__bf16*)alloc((size_t)NN * 768 * 2);  // bf16 q|k|v
    float*  skipb = (float*)alloc((size_t)NN * 256 * 4);   // fp32 skip
    float*  accb  = (float*)alloc((size_t)NN * 256 * 4);
    float*  zb    = (float*)alloc((size_t)NN * 4 * 4);
    float*  mb    = (float*)alloc((size_t)NN * 4 * 4);
    float*  sc    = (float*)alloc((size_t)NE * 4 * 4);

    // ---- weight prep: transpose+convert to bf16 [N][K], pack biases ----
    struct LP { int base, din, HC, H; };
    const LP LPs[3] = { {6, 64, 256, 4}, {17, 256, 256, 4}, {28, 256, 64, 1} };
    __bf16* WTs[3]  = { WT1, WT2, WT3 };
    float*  bias[3] = { bia1, bia2, bia3 };
    for (int l = 0; l < 3; ++l) {
        int din = LPs[l].din, HC = LPs[l].HC;
        const int widx[4] = { LPs[l].base + 0, LPs[l].base + 2, LPs[l].base + 4, LPs[l].base + 7 };
        const int bidx[4] = { LPs[l].base + 1, LPs[l].base + 3, LPs[l].base + 5, LPs[l].base + 8 };
        for (int s = 0; s < 4; ++s) {
            int n = din * HC;
            k_wt_convert<<<CDIV(n, 256), 256, 0, stream>>>(
                (const float*)d_in[widx[s]], WTs[l] + (size_t)s * HC * din, din, HC);
            k_copy_f32<<<CDIV(HC, 256), 256, 0, stream>>>(
                (const float*)d_in[bidx[s]], bias[l] + s * HC, HC);
        }
    }
    k_wt_convert<<<CDIV(64 * 128, 256), 256, 0, stream>>>(w1, WTm, 128, 64);
    k_enc_table<<<CDIV(16 * 64, 256), 256, 0, stream>>>(te, encw, encb, tab);

    // ---- node encoder (writes bf16 h directly) ----
    k_encode<<<CDIV(NN * 64, 256), 256, 0, stream>>>(x, tab, hbf);

    const int EB = CDIV(NE, 8);   // edge kernels: 8 waves/block, wave per edge
    const int NB = CDIV(NN, 8);

    // ---- layer 1: din=64, HC=256, H=4, no residual.  -> hB (+hbf) ----
    {
        const float* we  = (const float*)d_in[12];
        const float* lnw = (const float*)d_in[15];
        const float* lnb = (const float*)d_in[16];
        k_gemm<64><<<CDIV((NN / 16) * (1024 / 16), 4), 128, 0, stream>>>(
            hbf, WT1, bia1, qkvb, skipb, NN, 1024, 768);
        k_init<<<CDIV(NN * 256, 256), 256, 0, stream>>>(mb, zb, accb, NN * 4, NN * 256);
        k_edge_scores<4, 256><<<EB, 256, 0, stream>>>(ei, ea, we, qkvb, sc, mb);
        k_edge_accum<4, 256><<<EB, 256, 0, stream>>>(ei, ea, we, qkvb, sc, mb, zb, accb);
        k_node_post<4, 256, false><<<NB, 256, 0, stream>>>(skipb, accb, zb, lnw, lnb, hA, hB, hbf);
    }
    // ---- layer 2: din=256, HC=256, H=4, residual (hB).  -> hA (+hbf) ----
    {
        const float* we  = (const float*)d_in[23];
        const float* lnw = (const float*)d_in[26];
        const float* lnb = (const float*)d_in[27];
        k_gemm<256><<<CDIV((NN / 16) * (1024 / 16), 4), 128, 0, stream>>>(
            hbf, WT2, bia2, qkvb, skipb, NN, 1024, 768);
        k_init<<<CDIV(NN * 256, 256), 256, 0, stream>>>(mb, zb, accb, NN * 4, NN * 256);
        k_edge_scores<4, 256><<<EB, 256, 0, stream>>>(ei, ea, we, qkvb, sc, mb);
        k_edge_accum<4, 256><<<EB, 256, 0, stream>>>(ei, ea, we, qkvb, sc, mb, zb, accb);
        k_node_post<4, 256, true><<<NB, 256, 0, stream>>>(skipb, accb, zb, lnw, lnb, hB, hA, hbf);
    }
    // ---- layer 3: din=256, HC=64, H=1, no residual.  -> hB (+hbf) ----
    {
        const float* we  = (const float*)d_in[34];
        const float* lnw = (const float*)d_in[37];
        const float* lnb = (const float*)d_in[38];
        k_gemm<256><<<CDIV((NN / 16) * (256 / 16), 4), 128, 0, stream>>>(
            hbf, WT3, bia3, qkvb, skipb, NN, 256, 192);
        k_init<<<CDIV(NN * 64, 256), 256, 0, stream>>>(mb, zb, accb, NN * 1, NN * 64);
        k_edge_scores<1, 64><<<EB, 256, 0, stream>>>(ei, ea, we, qkvb, sc, mb);
        k_edge_accum<1, 64><<<EB, 256, 0, stream>>>(ei, ea, we, qkvb, sc, mb, zb, accb);
        k_node_post<1, 64, false><<<NB, 256, 0, stream>>>(skipb, accb, zb, lnw, lnb, hA, hB, hbf);
    }
    // ---- edge MLP on bf16 h: -> out [NE,3] ----
    k_mlp<<<CDIV(NE / 16, 8), 256, 0, stream>>>(ei, hbf, WTm, b1, w2, b2, out);
}
